// CorrelationModule_72730976190657
// MI455X (gfx1250) — compile-verified, matrix operand order checked by reading, not run on hardware
//
#include <hip/hip_runtime.h>
#include <hip/hip_bf16.h>

// ---------------------------------------------------------------------------
// CDNA5 (gfx1250) attention module, fixed problem size:
//   B=8, C=384, O=512, N=48*48=2304
// q/k/v 1x1-conv projections + softmax(q k^T/sqrt(O)) + v*attn^T + Wo proj.
// All GEMMs: v_wmma_f32_16x16x32_f16. Flash attention with K/V tiles
// double-buffered in LDS via GLOBAL_LOAD_ASYNC_TO_LDS_B128 (ASYNCcnt).
// ---------------------------------------------------------------------------

typedef __attribute__((ext_vector_type(16))) _Float16 v16h;
typedef __attribute__((ext_vector_type(8)))  _Float16 v8h;
typedef __attribute__((ext_vector_type(8)))  float    v8f;
typedef __attribute__((ext_vector_type(4)))  float    f32x4;
typedef __attribute__((ext_vector_type(4)))  int      v4i;

#define ASG __attribute__((address_space(1)))
#define ASL __attribute__((address_space(3)))
typedef ASG v4i* gv4i_p;   // addrspace(1) int4*
typedef ASL v4i* lv4i_p;   // addrspace(3) int4*

#if defined(__has_builtin)
#if __has_builtin(__builtin_amdgcn_global_load_async_to_lds_b128)
#define HAVE_ASYNC_LDS 1
#endif
#endif

constexpr int CB = 8;      // batch
constexpr int CC = 384;    // in channels
constexpr int CO = 512;    // out channels
constexpr int CN = 2304;   // tokens (48*48)

union Frag16 {
    v16h v;
    struct { v8h lo, hi; };
};

// Load one 16x32 f16 A/B WMMA fragment from a K-contiguous matrix
// (global or LDS). Lane element i -> k = k0 + (i<8?0:16) + 8*half + (i&7).
__device__ inline v16h load_frag(const _Float16* base, int ld, int row, int k0, int half) {
    Frag16 f;
    const _Float16* p = base + (size_t)row * (size_t)ld + (size_t)(k0 + 8 * half);
    f.lo = *(const v8h*)(p);
    f.hi = *(const v8h*)(p + 16);
    return f.v;
}

__device__ inline v8f wmma_f16(v16h a, v16h b, v8f c) {
    return __builtin_amdgcn_wmma_f32_16x16x32_f16(false, a, false, b, (short)0, c, false, false);
}

// 16-byte global -> LDS copy (async when available).
// loff = byte offset inside the dynamic LDS segment (== AS(3) address).
__device__ inline void cp16(_Float16* lp, unsigned loff, const _Float16* g) {
#ifdef HAVE_ASYNC_LDS
    __builtin_amdgcn_global_load_async_to_lds_b128(
        (gv4i_p)(unsigned long long)g, (lv4i_p)(unsigned long long)loff, 0, 0);
    (void)lp;
#else
    *(f32x4*)lp = *(const f32x4*)g;
    (void)loff;
#endif
}

__device__ inline void wait_stage() {
#ifdef HAVE_ASYNC_LDS
#if __has_builtin(__builtin_amdgcn_s_wait_asynccnt)
    __builtin_amdgcn_s_wait_asynccnt(0);
#else
    asm volatile("s_wait_asynccnt 0x0" ::: "memory");
#endif
#endif
}

// ---------------------------------------------------------------------------
// Kernel 1: x[B,C,N] f32 -> xh[B,N,C] f16 (tiled transpose through LDS)
// ---------------------------------------------------------------------------
__global__ void __launch_bounds__(256)
k_convert_x(const float* __restrict__ x, _Float16* __restrict__ xh) {
    __shared__ float tile[32][33];
    const int b  = blockIdx.z;
    const int c0 = blockIdx.x * 32;
    const int n0 = blockIdx.y * 32;
    const int tx = threadIdx.x;
    const int ty = threadIdx.y;
    const float* xb = x + (size_t)b * CC * CN;
    for (int i = 0; i < 32; i += 8)
        tile[ty + i][tx] = xb[(size_t)(c0 + ty + i) * CN + (n0 + tx)];
    __syncthreads();
    _Float16* xhb = xh + (size_t)b * CN * CC;
    for (int i = 0; i < 32; i += 8)
        xhb[(size_t)(n0 + ty + i) * CC + (c0 + tx)] = (_Float16)tile[tx][ty + i];
}

// ---------------------------------------------------------------------------
// Kernel 2: f32 -> f16 elementwise (weights)
// ---------------------------------------------------------------------------
__global__ void __launch_bounds__(256)
k_convert_w(const float* __restrict__ w, _Float16* __restrict__ wh, int n) {
    int i = blockIdx.x * 256 + threadIdx.x;
    if (i < n) wh[i] = (_Float16)w[i];
}

// ---------------------------------------------------------------------------
// Kernel 3: projection GEMM (fully unrolled k-loop, KS = 384/32 = 12).
// out[n,o] = (sum_c Xh[n,c]*Wh[o,c] + bias[o]) * scale
// transposed==0 -> [B,N,O];  transposed==1 -> [B,O,N] (V).
// ---------------------------------------------------------------------------
__global__ void __launch_bounds__(128)
k_proj(const _Float16* __restrict__ Xh, const _Float16* __restrict__ Wh,
       const float* __restrict__ bias, _Float16* __restrict__ out,
       float scale, int transposed) {
    const int b     = blockIdx.z;
    const int nbase = blockIdx.x * 64;
    const int obase = blockIdx.y * 64;
    const int lane  = threadIdx.x & 31;
    const int wv    = threadIdx.x >> 5;
    const int col   = lane & 15;
    const int half  = lane >> 4;

    const _Float16* Xb = Xh + (size_t)b * CN * CC;
    const int nrow = nbase + wv * 16;

    v8f acc[4] = {};
#pragma unroll
    for (int ks = 0; ks < CC / 32; ++ks) {
        v16h a = load_frag(Xb, CC, nrow + col, ks * 32, half);
#pragma unroll
        for (int ct = 0; ct < 4; ++ct) {
            v16h bf = load_frag(Wh, CC, obase + ct * 16 + col, ks * 32, half);
            acc[ct] = wmma_f16(a, bf, acc[ct]);
        }
    }

#pragma unroll
    for (int ct = 0; ct < 4; ++ct) {
        const int o = obase + ct * 16 + col;
        const float bs = bias[o];
#pragma unroll
        for (int r = 0; r < 8; ++r) {
            const int n = nrow + r + 8 * half;
            const float vv = (acc[ct][r] + bs) * scale;
            if (transposed) out[((size_t)b * CO + o) * CN + n] = (_Float16)vv;
            else            out[((size_t)b * CN + n) * CO + o] = (_Float16)vv;
        }
    }
}

// ---------------------------------------------------------------------------
// Kernel 4: flash attention. 64 query rows / block, 512 thr = 16 waves.
// Q strip lives in registers (16 frags/wave). K-tile (64x512) and V-slice
// (512x64) double-buffered in LDS, staged with async-to-LDS while the
// previous tile is being computed. Wave w: qt = w&3, sub = w>>2.
// ---------------------------------------------------------------------------
#define QSTR 520            // K LDS row stride: 512 + 8 f16 pad
#define VSTR 72             // V/P LDS row stride: 64 + 8 f16 pad
#define SSTR 66             // S LDS row stride: 64 + 2 f32 pad
#define KOFF0 0u
#define KOFF1 66560u        // 64*520*2
#define VOFF0 133120u
#define VOFF1 206848u       // + 512*72*2
#define SOFF  280576u
#define POFF  297472u       // + 64*66*4
#define MROW  306688u       // + 64*72*2
#define LROW  306944u
#define SROW  307200u
#define REDO  307456u       // red[64][8] f32
#define FLASH_LDS 309504u

// stage 64 rows x 512 f16 (row stride ldg) -> LDS rows of stride QSTR
__device__ inline void stage64x512(const _Float16* g, int ldg, char* sm,
                                   unsigned loff0, int tid) {
    const int row = tid >> 3;
    const int seg = (tid & 7) * 64;
    const _Float16* gp = g + (size_t)row * ldg + seg;
    const unsigned lo = loff0 + (unsigned)(row * QSTR + seg) * 2u;
    _Float16* lp = (_Float16*)(sm + lo);
#pragma unroll
    for (int j = 0; j < 8; ++j)
        cp16(lp + j * 8, lo + (unsigned)j * 16u, gp + j * 8);
}

// stage 512 rows x 64 f16 (row stride ldg) -> LDS rows of stride VSTR
__device__ inline void stage512x64(const _Float16* g, int ldg, char* sm,
                                   unsigned loff0, int tid) {
    const _Float16* gp = g + (size_t)tid * ldg;
    const unsigned lo = loff0 + (unsigned)(tid * VSTR) * 2u;
    _Float16* lp = (_Float16*)(sm + lo);
#pragma unroll
    for (int j = 0; j < 8; ++j)
        cp16(lp + j * 8, lo + (unsigned)j * 16u, gp + j * 8);
}

__global__ void __launch_bounds__(512)
k_flash(const _Float16* __restrict__ Qh, const _Float16* __restrict__ Kh,
        const _Float16* __restrict__ Vh, _Float16* __restrict__ Yh) {
    extern __shared__ char smem[];
    float*    Ss   = (float*)   (smem + SOFF);
    _Float16* Ps   = (_Float16*)(smem + POFF);
    float*    mrow = (float*)   (smem + MROW);
    float*    lrow = (float*)   (smem + LROW);
    float*    srow = (float*)   (smem + SROW);
    float*    red  = (float*)   (smem + REDO);

    const int tid  = threadIdx.x;
    const int lane = tid & 31;
    const int wv   = tid >> 5;       // 0..15
    const int col  = lane & 15;
    const int half = lane >> 4;
    const int qt   = wv & 3;         // query subtile (16 rows)
    const int sub  = wv >> 2;        // key subtile / o-slice
    const int row8 = tid >> 3;       // softmax row (0..63)
    const int part = tid & 7;        // softmax column-chunk (8 cols each)

    const int b     = blockIdx.y;
    const int qbase = blockIdx.x * 64;

    const _Float16* Qb = Qh + (size_t)b * CN * CO;
    const _Float16* Kb = Kh + (size_t)b * CN * CO;
    const _Float16* Vb = Vh + (size_t)b * CO * CN;

    // ---- Q strip into registers: 16 fragments (this wave's 16 q rows) -----
    v16h qf[16];
#pragma unroll
    for (int ks = 0; ks < 16; ++ks)
        qf[ks] = load_frag(Qb, CO, qbase + qt * 16 + col, ks * 32, half);

    if (tid < 64) { mrow[tid] = -3.0e38f; lrow[tid] = 0.0f; }

    // ---- prologue: stage tile 0 into buffer 0 -----------------------------
    stage64x512(Kb, CO, smem, KOFF0, tid);
    stage512x64(Vb, CN, smem, VOFF0, tid);

    v8f acc[8] = {};
    constexpr int ntiles = CN / 64;  // 36

    for (int kt = 0; kt < ntiles; ++kt) {
        wait_stage();
        __syncthreads();             // staged buffer (kt&1) ready everywhere

        const unsigned kcur = (kt & 1) ? KOFF1 : KOFF0;
        const unsigned vcur = (kt & 1) ? VOFF1 : VOFF0;
        _Float16* Ksc = (_Float16*)(smem + kcur);
        _Float16* Vsc = (_Float16*)(smem + vcur);

        // kick off async staging of the next tile into the other buffer
        if (kt + 1 < ntiles) {
            stage64x512(Kb + (size_t)(kt + 1) * 64 * CO, CO, smem,
                        (kt & 1) ? KOFF0 : KOFF1, tid);
            stage512x64(Vb + (kt + 1) * 64, CN, smem,
                        (kt & 1) ? VOFF0 : VOFF1, tid);
        }

        // ---- S subtile (16q x 16k), fully unrolled over K=512 -------------
        v8f s = {};
#pragma unroll
        for (int ks = 0; ks < 16; ++ks) {
            v16h bk = load_frag(Ksc, QSTR, sub * 16 + col, ks * 32, half);
            s = wmma_f16(qf[ks], bk, s);
        }
#pragma unroll
        for (int r = 0; r < 8; ++r)
            Ss[(qt * 16 + r + 8 * half) * SSTR + sub * 16 + col] = s[r];
        __syncthreads();

        // ---- parallel online softmax (8 threads per row) ------------------
        const float* sp = Ss + row8 * SSTR + part * 8;
        {   // phase A: partial max
            float pm = -3.0e38f;
#pragma unroll
            for (int j = 0; j < 8; ++j) pm = fmaxf(pm, sp[j]);
            red[row8 * 8 + part] = pm;
        }
        __syncthreads();
        if (tid < 64) {  // phase B: combine max, compute rescale factor
            const float m_old = mrow[tid];
            float tm = m_old;
#pragma unroll
            for (int j = 0; j < 8; ++j) tm = fmaxf(tm, red[tid * 8 + j]);
            mrow[tid] = tm;
            srow[tid] = __expf(m_old - tm);
        }
        __syncthreads();
        {   // phase C: exponentiate, write P, partial sums
            const float mn = mrow[row8];
            _Float16* pp = Ps + row8 * VSTR + part * 8;
            float sum = 0.0f;
#pragma unroll
            for (int j = 0; j < 8; ++j) {
                float e = __expf(sp[j] - mn);
                pp[j] = (_Float16)e;
                sum += e;
            }
            red[row8 * 8 + part] = sum;
        }
        __syncthreads();
        if (tid < 64) {  // phase D: fold partial sums into running l
            float sm2 = 0.0f;
#pragma unroll
            for (int j = 0; j < 8; ++j) sm2 += red[tid * 8 + j];
            lrow[tid] = lrow[tid] * srow[tid] + sm2;
        }
        __syncthreads();

        // ---- rescale accumulators, then P*V -------------------------------
#pragma unroll
        for (int ct = 0; ct < 8; ++ct)
#pragma unroll
            for (int r = 0; r < 8; ++r)
                acc[ct][r] *= srow[qt * 16 + r + 8 * half];

#pragma unroll
        for (int ct = 0; ct < 8; ++ct) {
            const int orow = sub * 128 + ct * 16 + col;
#pragma unroll
            for (int ks = 0; ks < 2; ++ks) {
                v16h ap = load_frag(Ps, VSTR, qt * 16 + col, ks * 32, half);
                v16h bv = load_frag(Vsc, VSTR, orow, ks * 32, half);
                acc[ct] = wmma_f16(ap, bv, acc[ct]);
            }
        }
        // next-iteration top barrier protects S/P and the staged buffers
    }

    // ---- epilogue: normalize, store Y as [B,N,O] ---------------------------
#pragma unroll
    for (int ct = 0; ct < 8; ++ct) {
        const int o = sub * 128 + ct * 16 + col;
#pragma unroll
        for (int r = 0; r < 8; ++r) {
            const int n = qbase + qt * 16 + r + 8 * half;
            const float inv = 1.0f / lrow[qt * 16 + r + 8 * half];
            Yh[((size_t)b * CN + n) * CO + o] = (_Float16)(acc[ct][r] * inv);
        }
    }
}

// ---------------------------------------------------------------------------
// Kernel 5: final projection (fully unrolled, KS = 512/32 = 16).
// out[b,p,n] = sum_o Wo[p,o] * Y[b,n,o] + bo[p]
// ---------------------------------------------------------------------------
__global__ void __launch_bounds__(128)
k_final(const _Float16* __restrict__ Woh, const _Float16* __restrict__ Yh,
        const float* __restrict__ bo, float* __restrict__ out) {
    const int b     = blockIdx.z;
    const int pbase = blockIdx.x * 64;
    const int nbase = blockIdx.y * 64;
    const int lane  = threadIdx.x & 31;
    const int wv    = threadIdx.x >> 5;
    const int col   = lane & 15;
    const int half  = lane >> 4;

    const _Float16* Yb = Yh + (size_t)b * CN * CO;
    const int prow = pbase + wv * 16;

    v8f acc[4] = {};
#pragma unroll
    for (int ks = 0; ks < CO / 32; ++ks) {
        v16h a = load_frag(Woh, CO, prow + col, ks * 32, half);
#pragma unroll
        for (int ct = 0; ct < 4; ++ct) {
            v16h bf = load_frag(Yb, CO, nbase + ct * 16 + col, ks * 32, half);
            acc[ct] = wmma_f16(a, bf, acc[ct]);
        }
    }

#pragma unroll
    for (int ct = 0; ct < 4; ++ct) {
        const int n = nbase + ct * 16 + col;
#pragma unroll
        for (int r = 0; r < 8; ++r) {
            const int p = prow + r + 8 * half;
            out[((size_t)b * CO + p) * CN + n] = acc[ct][r] + bo[p];
        }
    }
}

// ---------------------------------------------------------------------------
// Host launcher
// ---------------------------------------------------------------------------
extern "C" void kernel_launch(void* const* d_in, const int* in_sizes, int n_in,
                              void* d_out, int out_size, void* d_ws, size_t ws_size,
                              hipStream_t stream) {
    (void)in_sizes; (void)n_in; (void)out_size; (void)ws_size;

    const int B = CB, C = CC, O = CO, N = CN;

    const float* x  = (const float*)d_in[0];
    const float* Wq = (const float*)d_in[1];
    const float* bq = (const float*)d_in[2];
    const float* Wk = (const float*)d_in[3];
    const float* bk = (const float*)d_in[4];
    const float* Wv = (const float*)d_in[5];
    const float* bv = (const float*)d_in[6];
    const float* Wo = (const float*)d_in[7];
    const float* bo = (const float*)d_in[8];
    float* out = (float*)d_out;

    char* ws = (char*)d_ws;
    size_t off = 0;
    auto carve = [&](size_t bytes) -> char* {
        char* p = ws + off;
        off = (off + bytes + 255) & ~(size_t)255;
        return p;
    };
    _Float16* xh  = (_Float16*)carve((size_t)B * N * C * 2);
    _Float16* Wqh = (_Float16*)carve((size_t)O * C * 2);
    _Float16* Wkh = (_Float16*)carve((size_t)O * C * 2);
    _Float16* Wvh = (_Float16*)carve((size_t)O * C * 2);
    _Float16* Woh = (_Float16*)carve((size_t)O * O * 2);
    _Float16* Qh  = (_Float16*)carve((size_t)B * N * O * 2);
    _Float16* Kh  = (_Float16*)carve((size_t)B * N * O * 2);
    _Float16* Vh  = (_Float16*)carve((size_t)B * O * N * 2);
    _Float16* Yh  = (_Float16*)carve((size_t)B * N * O * 2);

    (void)hipFuncSetAttribute((const void*)k_flash,
                              hipFuncAttributeMaxDynamicSharedMemorySize,
                              (int)FLASH_LDS);

    k_convert_x<<<dim3(C / 32, N / 32, B), dim3(32, 8), 0, stream>>>(x, xh);

    k_convert_w<<<dim3((O * C + 255) / 256), dim3(256), 0, stream>>>(Wq, Wqh, O * C);
    k_convert_w<<<dim3((O * C + 255) / 256), dim3(256), 0, stream>>>(Wk, Wkh, O * C);
    k_convert_w<<<dim3((O * C + 255) / 256), dim3(256), 0, stream>>>(Wv, Wvh, O * C);
    k_convert_w<<<dim3((O * O + 255) / 256), dim3(256), 0, stream>>>(Wo, Woh, O * O);

    const float scale = 1.0f / sqrtf((float)O);
    dim3 pg(N / 64, O / 64, B);
    k_proj<<<pg, dim3(128), 0, stream>>>(xh, Wqh, bq, Qh, scale, 0);
    k_proj<<<pg, dim3(128), 0, stream>>>(xh, Wkh, bk, Kh, 1.0f, 0);
    k_proj<<<pg, dim3(128), 0, stream>>>(xh, Wvh, bv, Vh, 1.0f, 1);

    k_flash<<<dim3(N / 64, B), dim3(512), FLASH_LDS, stream>>>(Qh, Kh, Vh, Yh);

    k_final<<<dim3(O / 64, N / 64, B), dim3(128), 0, stream>>>(Woh, Yh, bo, out);
}
